// AttentionWithBias_85237920956531
// MI455X (gfx1250) — matrix operand order repaired and checked by reading, not compiled, not run on hardware
//
#include <hip/hip_runtime.h>
#include <math.h>

// ---------------- types ----------------
typedef __bf16 bf16;
typedef __attribute__((ext_vector_type(16))) __bf16 v16bf;
typedef __attribute__((ext_vector_type(8)))  float  v8f;
typedef int v4i __attribute__((vector_size(16)));   // matches builtin prototype V4i

union FragBF {            // one WMMA 16-bit A/B fragment: 16 bf16 per lane = 8 VGPRs
    v16bf v;
    uint4 u[2];
};

__device__ __forceinline__ bf16 f2bf(float f) {
    union { float f; unsigned u; } a; a.f = f;
    unsigned r = (a.u + 0x7FFFu + ((a.u >> 16) & 1u)) >> 16;   // round-to-nearest-even
    union { unsigned short s; bf16 b; } o; o.s = (unsigned short)r;
    return o.b;
}
__device__ __forceinline__ unsigned short f2bfraw(float f) {
    union { float f; unsigned u; } a; a.f = f;
    return (unsigned short)((a.u + 0x7FFFu + ((a.u >> 16) & 1u)) >> 16);
}

// Problem constants
#define BB 2
#define TT 2048
#define DD 1024
#define HH 16
#define HDD 64
#define MM (BB * TT)       // 4096

// CDNA5 async global->LDS path (ASYNCcnt), guarded so either toolchain compiles.
#if defined(__has_builtin)
#if __has_builtin(__builtin_amdgcn_global_load_async_to_lds_b128) && \
    __has_builtin(__builtin_amdgcn_s_wait_asynccnt)
#define USE_ASYNC_LDS 1
#endif
#endif
#ifndef USE_ASYNC_LDS
#define USE_ASYNC_LDS 0
#endif

#if USE_ASYNC_LDS
__device__ __forceinline__ void async_b128(const void* g, void* l) {
    __builtin_amdgcn_global_load_async_to_lds_b128(
        (__attribute__((address_space(1))) v4i*)(g),
        (__attribute__((address_space(3))) v4i*)(l), 0, 0);
}
#endif

// ---------------- f32 -> bf16 conversion ----------------
__global__ __launch_bounds__(256) void cvt_kernel(const float* __restrict__ src,
                                                  bf16* __restrict__ dst, int n) {
    int i = (blockIdx.x * 256 + threadIdx.x) * 4;
    if (i + 3 < n) {
        float4 f = *(const float4*)(src + i);
        dst[i + 0] = f2bf(f.x);
        dst[i + 1] = f2bf(f.y);
        dst[i + 2] = f2bf(f.z);
        dst[i + 3] = f2bf(f.w);
    }
}

// ---------------- GEMM: out = A[M,K] @ W[N,K]^T + bias ----------------
// mode 0: Q  -> bf16 [B,H,T,HD], scaled by 1/sqrt(HD)
// mode 1: K  -> bf16 [B,H,T,HD]
// mode 2: V  -> bf16 [B,H,HD,T]   (transposed for PV B-fragments)
// mode 3: O  -> f32  [M,D] (final output)
__global__ __launch_bounds__(256) void gemm_kernel(const bf16* __restrict__ A,
                                                   const bf16* __restrict__ W,
                                                   const float* __restrict__ bias,
                                                   bf16* __restrict__ outB,
                                                   float* __restrict__ outF,
                                                   int mode) {
    // double-buffered 64(M) x 64(K) A tile and 128(N) x 64(K) B tile
    // rows padded 64->72 elements (144B stride, 16B aligned, bank-spread)
    __shared__ bf16 As[2][64][72];
    __shared__ bf16 Bs[2][128][72];

    const int tid  = threadIdx.x;
    const int lane = tid & 31;
    const int w    = tid >> 5;
    const int wr   = w >> 2;       // 0..1
    const int wc   = w & 3;        // 0..3
    const int mB   = blockIdx.y * 64;
    const int nB   = blockIdx.x * 128;
    const int l15  = lane & 15;
    const int hi   = (lane >> 4) & 1;

    v8f acc[2][2] = {};

    const int arow = tid >> 2;                 // 0..63
    const int acg  = (tid & 3) * 8;            // 0,8,16,24 (+32 second chunk)
    const int brow = tid >> 1;                 // 0..127
    const int bcg  = (tid & 1) * 32;           // 0,32 (4 chunks of 8)

    const bf16* aBase = A + (size_t)(mB + arow) * DD;
    const bf16* bBase = W + (size_t)(nB + brow) * DD;

    const int akb = hi ? 8 : 0;    // A layout: lanes>=16 start at K=8 / K=24
    const int bkb = hi ? 8 : 0;    // B layout: lanes>=16 start at K=8 / K=24

    int p = 0;
#if USE_ASYNC_LDS
    // prologue: stream tile 0 into buffer 0 via the async-LDS path (ASYNCcnt)
    {
        async_b128(aBase + acg,       &As[0][arow][acg]);
        async_b128(aBase + acg + 32,  &As[0][arow][acg + 32]);
        async_b128(bBase + bcg,       &Bs[0][brow][bcg]);
        async_b128(bBase + bcg + 8,   &Bs[0][brow][bcg + 8]);
        async_b128(bBase + bcg + 16,  &Bs[0][brow][bcg + 16]);
        async_b128(bBase + bcg + 24,  &Bs[0][brow][bcg + 24]);
    }
#endif

    for (int k0 = 0; k0 < DD; k0 += 64) {
#if USE_ASYNC_LDS
        const int kn = k0 + 64;
        if (kn < DD) {   // issue next tile while current streams/computes
            const int q = p ^ 1;
            async_b128(aBase + kn + acg,       &As[q][arow][acg]);
            async_b128(aBase + kn + acg + 32,  &As[q][arow][acg + 32]);
            async_b128(bBase + kn + bcg,       &Bs[q][brow][bcg]);
            async_b128(bBase + kn + bcg + 8,   &Bs[q][brow][bcg + 8]);
            async_b128(bBase + kn + bcg + 16,  &Bs[q][brow][bcg + 16]);
            async_b128(bBase + kn + bcg + 24,  &Bs[q][brow][bcg + 24]);
            __builtin_amdgcn_s_wait_asynccnt(6);   // tile k0 landed (6 in flight = next tile)
        } else {
            __builtin_amdgcn_s_wait_asynccnt(0);
        }
        __syncthreads();
#else
        uint4 a0 = *(const uint4*)(aBase + k0 + acg);
        uint4 a1 = *(const uint4*)(aBase + k0 + acg + 32);
        uint4 b0 = *(const uint4*)(bBase + k0 + bcg);
        uint4 b1 = *(const uint4*)(bBase + k0 + bcg + 8);
        uint4 b2 = *(const uint4*)(bBase + k0 + bcg + 16);
        uint4 b3 = *(const uint4*)(bBase + k0 + bcg + 24);
        if (k0 + 64 < DD) {
            __builtin_prefetch(aBase + k0 + 64 + acg, 0, 0);
            __builtin_prefetch(bBase + k0 + 64 + bcg, 0, 0);
        }
        __syncthreads();
        *(uint4*)(&As[p][arow][acg])      = a0;
        *(uint4*)(&As[p][arow][acg + 32]) = a1;
        *(uint4*)(&Bs[p][brow][bcg])      = b0;
        *(uint4*)(&Bs[p][brow][bcg + 8])  = b1;
        *(uint4*)(&Bs[p][brow][bcg + 16]) = b2;
        *(uint4*)(&Bs[p][brow][bcg + 24]) = b3;
        __syncthreads();
#endif
        for (int ks = 0; ks < 2; ++ks) {
            const int ko = ks * 32;
            FragBF a[2], b[2];
            for (int mt = 0; mt < 2; ++mt) {
                int row = wr * 32 + mt * 16 + l15;
                a[mt].u[0] = *(const uint4*)(&As[p][row][ko + akb]);
                a[mt].u[1] = *(const uint4*)(&As[p][row][ko + akb + 16]);
            }
            for (int nt = 0; nt < 2; ++nt) {
                int col = wc * 32 + nt * 16 + l15;
                b[nt].u[0] = *(const uint4*)(&Bs[p][col][ko + bkb]);
                b[nt].u[1] = *(const uint4*)(&Bs[p][col][ko + bkb + 16]);
            }
            for (int mt = 0; mt < 2; ++mt)
                for (int nt = 0; nt < 2; ++nt)
                    acc[mt][nt] = __builtin_amdgcn_wmma_f32_16x16x32_bf16(
                        false, a[mt].v, false, b[nt].v, (short)0, acc[mt][nt], false, false);
        }
        __syncthreads();   // all reads of buffer p done before it is refilled
        p ^= 1;
    }

    // epilogue: bias + scatter per mode
    for (int mt = 0; mt < 2; ++mt) {
        for (int nt = 0; nt < 2; ++nt) {
            int gn = nB + wc * 32 + nt * 16 + l15;
            float bv = bias[gn];
            if (mode == 2) {
                // V^T [B,H,HD,T]: a lane's 8 rows are 8 consecutive t -> one b128 store
                union { unsigned short s[8]; uint4 u; } pk;
                int gm0 = mB + wr * 32 + mt * 16 + hi * 8;     // multiple of 8
                for (int r = 0; r < 8; ++r) pk.s[r] = f2bfraw(acc[mt][nt][r] + bv);
                int bb = gm0 >> 11, t0 = gm0 & (TT - 1);
                int h  = gn >> 6,  d  = gn & (HDD - 1);
                *(uint4*)(&outB[(((size_t)(bb * HH + h)) * HDD + d) * TT + t0]) = pk.u;
            } else {
                for (int r = 0; r < 8; ++r) {
                    int gm = mB + wr * 32 + mt * 16 + r + hi * 8;
                    float val = acc[mt][nt][r] + bv;
                    if (mode == 3) {
                        outF[(size_t)gm * DD + gn] = val;
                    } else {
                        int bb = gm >> 11, t = gm & (TT - 1);
                        int h  = gn >> 6,  d = gn & (HDD - 1);
                        float sv = (mode == 0) ? val * 0.125f : val;  // 1/sqrt(64)
                        outB[(((size_t)(bb * HH + h)) * TT + t) * HDD + d] = f2bf(sv);
                    }
                }
            }
        }
    }
}

// ---------------- flash attention: softmax(Q K^T + bias) V ----------------
// One wave per 16-query tile; 32 keys per iteration; bias streamed once (NT hint).
__global__ __launch_bounds__(256) void attn_kernel(const bf16* __restrict__ Q,
                                                   const bf16* __restrict__ K,
                                                   const bf16* __restrict__ VT,
                                                   const float* __restrict__ bias,
                                                   bf16* __restrict__ out) {
    __shared__ bf16 pbuf[8][16][40];   // per-wave P transpose scratch (C-layout -> A-frag)

    const int tid = threadIdx.x, lane = tid & 31, wid = tid >> 5;
    const int l15 = lane & 15, hi = (lane >> 4) & 1;
    const int gw = blockIdx.x * 8 + wid;    // 4096 waves = B*H*(T/16)
    const int qt = gw & 127;
    const int bh = gw >> 7;                  // b*H + h
    const int qbase = qt * 16;

    const bf16*  Qh = Q    + (size_t)bh * TT * HDD;
    const bf16*  Kh = K    + (size_t)bh * TT * HDD;
    const bf16*  Vh = VT   + (size_t)bh * HDD * TT;
    const float* Bh = bias + (size_t)bh * TT * TT;

    // Q A-fragments (2 frags cover K-dim HD=64), resident for the whole loop
    FragBF qf[2];
    {
        int m  = qbase + l15;
        int kb = hi ? 8 : 0;
        for (int f = 0; f < 2; ++f) {
            qf[f].u[0] = *(const uint4*)(Qh + (size_t)m * HDD + f * 32 + kb);
            qf[f].u[1] = *(const uint4*)(Qh + (size_t)m * HDD + f * 32 + kb + 16);
        }
    }

    float mrow[8], lrow[8];
    v8f O[4] = {};
    for (int r = 0; r < 8; ++r) { mrow[r] = -3.0e38f; lrow[r] = 0.f; }

    for (int j0 = 0; j0 < TT; j0 += 32) {
        // ---- scores for 32 keys (two 16-key WMMA N-tiles) ----
        v8f s[2];
        for (int sub = 0; sub < 2; ++sub) {
            int jb = j0 + sub * 16;
            FragBF kf[2];
            int n  = jb + l15;
            int kb = hi ? 8 : 0;
            for (int f = 0; f < 2; ++f) {
                kf[f].u[0] = *(const uint4*)(Kh + (size_t)n * HDD + f * 32 + kb);
                kf[f].u[1] = *(const uint4*)(Kh + (size_t)n * HDD + f * 32 + kb + 16);
            }
            v8f t = {};
            t = __builtin_amdgcn_wmma_f32_16x16x32_bf16(false, qf[0].v, false, kf[0].v,
                                                        (short)0, t, false, false);
            t = __builtin_amdgcn_wmma_f32_16x16x32_bf16(false, qf[1].v, false, kf[1].v,
                                                        (short)0, t, false, false);
            for (int r = 0; r < 8; ++r) {   // single-use stream: non-temporal
                int gm = qbase + r + hi * 8;
                t[r] += __builtin_nontemporal_load(&Bh[(size_t)gm * TT + jb + l15]);
            }
            s[sub] = t;
        }

        // ---- online softmax (row = VGPR index, reduce across 16 lanes) ----
        for (int r = 0; r < 8; ++r) {
            float cm = fmaxf(s[0][r], s[1][r]);
            for (int off = 8; off >= 1; off >>= 1)
                cm = fmaxf(cm, __shfl_xor(cm, off, 16));
            float mnew  = fmaxf(mrow[r], cm);
            float alpha = __expf(mrow[r] - mnew);
            float p0 = __expf(s[0][r] - mnew);
            float p1 = __expf(s[1][r] - mnew);
            float rs = p0 + p1;
            for (int off = 8; off >= 1; off >>= 1)
                rs += __shfl_xor(rs, off, 16);
            lrow[r] = lrow[r] * alpha + rs;
            mrow[r] = mnew;
            for (int nt = 0; nt < 4; ++nt) O[nt][r] *= alpha;
            s[0][r] = p0; s[1][r] = p1;
        }

        // ---- transpose P (C-layout) -> A-fragment via LDS ----
        for (int sub = 0; sub < 2; ++sub)
            for (int r = 0; r < 8; ++r)
                pbuf[wid][r + hi * 8][sub * 16 + l15] = f2bf(s[sub][r]);
        __syncthreads();
        FragBF pa;
        {
            int kb = hi ? 8 : 0;
            pa.u[0] = *(const uint4*)(&pbuf[wid][l15][kb]);
            pa.u[1] = *(const uint4*)(&pbuf[wid][l15][kb + 16]);
        }
        __syncthreads();

        // ---- O += P @ V  (V^T layout gives contiguous B-fragments) ----
        for (int nt = 0; nt < 4; ++nt) {
            FragBF vf;
            int d  = nt * 16 + l15;
            int kb = hi ? 8 : 0;
            vf.u[0] = *(const uint4*)(Vh + (size_t)d * TT + j0 + kb);
            vf.u[1] = *(const uint4*)(Vh + (size_t)d * TT + j0 + kb + 16);
            O[nt] = __builtin_amdgcn_wmma_f32_16x16x32_bf16(false, pa.v, false, vf.v,
                                                            (short)0, O[nt], false, false);
        }
    }

    // ---- normalize and store attn output as bf16 [M, D] for the Wo GEMM ----
    int b = bh >> 4, h = bh & 15;
    for (int nt = 0; nt < 4; ++nt) {
        for (int r = 0; r < 8; ++r) {
            int grow = b * TT + qbase + r + hi * 8;
            int gcol = h * HDD + nt * 16 + l15;
            out[(size_t)grow * DD + gcol] = f2bf(O[nt][r] / lrow[r]);
        }
    }
}

// ---------------- launcher ----------------
extern "C" void kernel_launch(void* const* d_in, const int* in_sizes, int n_in,
                              void* d_out, int out_size, void* d_ws, size_t ws_size,
                              hipStream_t stream) {
    (void)in_sizes; (void)n_in; (void)out_size; (void)ws_size;
    const float* x  = (const float*)d_in[0];
    const float* ab = (const float*)d_in[1];
    const float* Wq = (const float*)d_in[2];
    const float* bq = (const float*)d_in[3];
    const float* Wk = (const float*)d_in[4];
    const float* bk = (const float*)d_in[5];
    const float* Wv = (const float*)d_in[6];
    const float* bv = (const float*)d_in[7];
    const float* Wo = (const float*)d_in[8];
    const float* bo = (const float*)d_in[9];

    char* ws = (char*)d_ws;
    bf16* xb   = (bf16*)(ws);                 // 8 MB; later reused as attn output
    bf16* wqb  = (bf16*)(ws +  8388608u);     // 2 MB each
    bf16* wkb  = (bf16*)(ws + 10485760u);
    bf16* wvb  = (bf16*)(ws + 12582912u);
    bf16* wob  = (bf16*)(ws + 14680064u);
    bf16* qbuf = (bf16*)(ws + 16777216u);     // 8 MB
    bf16* kbuf = (bf16*)(ws + 25165824u);     // 8 MB
    bf16* vbuf = (bf16*)(ws + 33554432u);     // 8 MB  (total 40 MB)

    cvt_kernel<<<4096, 256, 0, stream>>>(x,  xb,  MM * DD);
    cvt_kernel<<<1024, 256, 0, stream>>>(Wq, wqb, DD * DD);
    cvt_kernel<<<1024, 256, 0, stream>>>(Wk, wkb, DD * DD);
    cvt_kernel<<<1024, 256, 0, stream>>>(Wv, wvb, DD * DD);
    cvt_kernel<<<1024, 256, 0, stream>>>(Wo, wob, DD * DD);

    dim3 gg(DD / 128, MM / 64);   // (8, 64)
    gemm_kernel<<<gg, 256, 0, stream>>>(xb, wqb, bq, qbuf, nullptr, 0);
    gemm_kernel<<<gg, 256, 0, stream>>>(xb, wkb, bk, kbuf, nullptr, 1);
    gemm_kernel<<<gg, 256, 0, stream>>>(xb, wvb, bv, vbuf, nullptr, 2);

    attn_kernel<<<512, 256, 0, stream>>>(qbuf, kbuf, vbuf, ab, xb);  // xb <- attn out

    gemm_kernel<<<gg, 256, 0, stream>>>(xb, wob, bo, nullptr, (float*)d_out, 3);
}
